// Res_GCN_12687333392402
// MI455X (gfx1250) — compile-verified, matrix operand order checked by reading, not run on hardware
//
#include <hip/hip_runtime.h>
#include <hip/hip_bf16.h>

typedef __attribute__((ext_vector_type(16))) _Float16 v16h;
typedef __attribute__((ext_vector_type(8)))  _Float16 v8h;
typedef __attribute__((ext_vector_type(8)))  float    v8f;

#define HD 128

// ---------------- degree / normalization ----------------
__global__ void k_deg_init(float* deg, int n) {
  int i = blockIdx.x * blockDim.x + threadIdx.x;
  if (i < n) deg[i] = 1.0f;                 // self loop contributes 1
}
__global__ void k_deg_count(const int* __restrict__ dst, float* deg, int e) {
  int i = blockIdx.x * blockDim.x + threadIdx.x;
  if (i < e) atomicAdd(&deg[dst[i]], 1.0f);
}
__global__ void k_dinv(float* deg, int n) {
  int i = blockIdx.x * blockDim.x + threadIdx.x;
  if (i < n) { float d = deg[i]; deg[i] = d > 0.0f ? rsqrtf(d) : 0.0f; }
}

// ---------------- WMMA GEMM: C[M x 128] = A[M x 128] * B[128 x 128] (+bias) ----
// transB == 0: B[k][n] = W[k*128 + n]   (weights stored [in, out])
// transB == 1: B[k][n] = W[n*128 + k]   (torch heads: y = x @ W.T, W = [out, in])
__global__ __launch_bounds__(128) void k_gemm128(
    const float* __restrict__ A, const float* __restrict__ W,
    const float* __restrict__ bias, float* __restrict__ C,
    int M, int transB)
{
  __shared__ _Float16 sW[HD * HD];          // stored [n][k] (column-major vs math B)
  int tid = threadIdx.x;
  for (int i = tid; i < HD * HD; i += 128) {
    int nn = i >> 7, kk = i & 127;
    float w = transB ? W[i] : W[kk * HD + nn];
    sW[i] = (_Float16)w;
  }
  __syncthreads();

  int wave = tid >> 5, lane = tid & 31;
  int m0 = blockIdx.x * 64 + wave * 16;     // 16-row strip per wave
  if (m0 >= M) return;                      // whole-wave uniform exit (after barrier)
  int ml = lane & 15, half = lane >> 4;

  v8f acc[8] = {};

  const float* arow = A + (size_t)(m0 + ml) * HD;

#pragma unroll
  for (int kc = 0; kc < 4; ++kc) {          // K = 128 in chunks of 32
    int kb = kc * 32 + half * 8;
    // A fragment: row (m0+ml), K = {kb..kb+7} U {kb+16..kb+23}
    v16h afrag;
#pragma unroll
    for (int j = 0; j < 8; ++j) {
      afrag[j]     = (_Float16)arow[kb + j];
      afrag[j + 8] = (_Float16)arow[kb + 16 + j];
    }
#pragma unroll
    for (int nt = 0; nt < 8; ++nt) {        // 8 column tiles of 16
      int n = nt * 16 + ml;
      const _Float16* bp = &sW[n * HD + kb];
      v8h b0 = *(const v8h*)(bp);           // ds_load_b128
      v8h b1 = *(const v8h*)(bp + 16);      // ds_load_b128
      v16h bfrag;
#pragma unroll
      for (int j = 0; j < 8; ++j) { bfrag[j] = b0[j]; bfrag[j + 8] = b1[j]; }
      acc[nt] = __builtin_amdgcn_wmma_f32_16x16x32_f16(
          false, afrag, false, bfrag, (short)0, acc[nt], false, false);
    }
  }

  // D layout: lane half selects M-octet, VGPR index is M%8, lane%16 is N
#pragma unroll
  for (int nt = 0; nt < 8; ++nt) {
    int n = nt * 16 + ml;
    float b = bias ? bias[n] : 0.0f;
#pragma unroll
    for (int v = 0; v < 8; ++v) {
      int m = m0 + v + half * 8;
      C[(size_t)m * HD + n] = acc[nt][v] + b;
    }
  }
}

// ---------------- aggregation ----------------
// agg[i,:] = t[i,:] * dinv[i]^2   (self loop; full overwrite -> no zeroing needed)
__global__ void k_selfloop(const float* __restrict__ t, const float* __restrict__ dinv,
                           float* __restrict__ agg, int n) {
  int gid = blockIdx.x * blockDim.x + threadIdx.x;
  if (gid >= n * 32) return;                // float4 granularity: 32 per row
  int row = gid >> 5;
  float d = dinv[row], s = d * d;
  float4 v = ((const float4*)t)[gid];
  v.x *= s; v.y *= s; v.z *= s; v.w *= s;
  ((float4*)agg)[gid] = v;
}

// one wave per edge: agg[dst,:] += t[src,:] * dinv[src]*dinv[dst]
__global__ void k_scatter(const int* __restrict__ src, const int* __restrict__ dst,
                          const float* __restrict__ dinv, const float* __restrict__ t,
                          float* __restrict__ agg, int e) {
  int w = blockIdx.x * (blockDim.x >> 5) + (threadIdx.x >> 5);
  if (w >= e) return;
  int lane = threadIdx.x & 31;
  int s = src[w], d = dst[w];
  float nrm = dinv[s] * dinv[d];
  float4 v = ((const float4*)(t + (size_t)s * HD))[lane];
  float* ap = agg + (size_t)d * HD + lane * 4;
  atomicAdd(ap + 0, v.x * nrm);
  atomicAdd(ap + 1, v.y * nrm);
  atomicAdd(ap + 2, v.z * nrm);
  atomicAdd(ap + 3, v.w * nrm);
}

__global__ void k_bias_act(float* __restrict__ x, const float* __restrict__ b,
                           int n, int relu) {
  int gid = blockIdx.x * blockDim.x + threadIdx.x;
  if (gid >= n * 32) return;
  int c4 = gid & 31;
  float4 v = ((float4*)x)[gid];
  float4 bb = ((const float4*)b)[c4];
  v.x += bb.x; v.y += bb.y; v.z += bb.z; v.w += bb.w;
  if (relu) {
    v.x = fmaxf(v.x, 0.f); v.y = fmaxf(v.y, 0.f);
    v.z = fmaxf(v.z, 0.f); v.w = fmaxf(v.w, 0.f);
  }
  ((float4*)x)[gid] = v;
}

__global__ void k_add(const float* __restrict__ a, const float* __restrict__ b,
                      float* __restrict__ o, int n) {
  int gid = blockIdx.x * blockDim.x + threadIdx.x;
  if (gid >= n * 32) return;
  float4 va = ((const float4*)a)[gid];
  float4 vb = ((const float4*)b)[gid];
  va.x += vb.x; va.y += vb.y; va.z += vb.z; va.w += vb.w;
  ((float4*)o)[gid] = va;
}

// out1[i,c] = (featL[i,:]/max(||featL[i,:]||,1e-12)) . Wh1[c,:] * 10 + bh1[c] * 10
__global__ void k_head1(const float* __restrict__ featL, const float* __restrict__ Wh1,
                        const float* __restrict__ bh1, float* __restrict__ out1, int n) {
  int node = blockIdx.x * (blockDim.x >> 5) + (threadIdx.x >> 5);
  if (node >= n) return;
  int lane = threadIdx.x & 31;
  float4 v = ((const float4*)(featL + (size_t)node * HD))[lane];
  float ss = v.x * v.x + v.y * v.y + v.z * v.z + v.w * v.w;
#pragma unroll
  for (int o = 16; o > 0; o >>= 1) ss += __shfl_xor(ss, o, 32);
  float inv = 10.0f / fmaxf(sqrtf(ss), 1e-12f);
#pragma unroll
  for (int c = 0; c < 4; ++c) {
    float4 w = ((const float4*)(Wh1 + c * HD))[lane];
    float dt = v.x * w.x + v.y * w.y + v.z * w.z + v.w * w.w;
#pragma unroll
    for (int o = 16; o > 0; o >>= 1) dt += __shfl_xor(dt, o, 32);
    if (lane == 0) out1[(size_t)node * 4 + c] = dt * inv + bh1[c] * 10.0f;
  }
}

extern "C" void kernel_launch(void* const* d_in, const int* in_sizes, int n_in,
                              void* d_out, int out_size, void* d_ws, size_t ws_size,
                              hipStream_t stream) {
  const float* x    = (const float*)d_in[0];
  const int*   ei   = (const int*)d_in[1];
  const float* W1   = (const float*)d_in[2];
  const float* b1   = (const float*)d_in[3];
  const float* W2   = (const float*)d_in[4];
  const float* b2   = (const float*)d_in[5];
  const float* W2u  = (const float*)d_in[6];
  const float* b2u  = (const float*)d_in[7];
  const float* Wh1  = (const float*)d_in[8];
  const float* bh1  = (const float*)d_in[9];
  const float* Wpca = (const float*)d_in[10];
  const float* bpca = (const float*)d_in[11];
  const float* Wbce = (const float*)d_in[12];
  const float* bbce = (const float*)d_in[13];

  int N = in_sizes[0] / HD;
  int E = in_sizes[1] / 2;
  const int* src = ei;
  const int* dst = ei + E;

  float* ws   = (float*)d_ws;
  float* dinv = ws;                               // N floats (deg -> dinv in place)
  float* t    = ws + N;                           // N*128 GEMM scratch
  float* h    = t + (size_t)N * HD;               // N*128 hidden, later reused as featU

  float* out   = (float*)d_out;
  float* out1  = out;                             // N*4
  float* opca  = out + (size_t)N * 4;             // N*128
  float* obce  = opca + (size_t)N * HD;           // N*128
  float* featL = obce + (size_t)N * HD;           // N*128 (4th output, written in place)

  int nb_n   = (N + 255) / 256;
  int nb_e   = (E + 255) / 256;
  int nb_nh4 = (N * 32 + 255) / 256;
  int nb_g   = (N + 63) / 64;
  int nb_ew  = (E + 7) / 8;
  int nb_nw  = (N + 7) / 8;

  // normalization coefficients
  k_deg_init <<<nb_n, 256, 0, stream>>>(dinv, N);
  k_deg_count<<<nb_e, 256, 0, stream>>>(dst, dinv, E);
  k_dinv     <<<nb_n, 256, 0, stream>>>(dinv, N);

  // conv1: h = relu(agg(x @ W1) + b1)
  k_gemm128  <<<nb_g, 128, 0, stream>>>(x, W1, nullptr, t, N, 0);
  k_selfloop <<<nb_nh4, 256, 0, stream>>>(t, dinv, h, N);
  k_scatter  <<<nb_ew, 256, 0, stream>>>(src, dst, dinv, t, h, E);
  k_bias_act <<<nb_nh4, 256, 0, stream>>>(h, b1, N, 1);

  // conv2 (label) -> featL (directly into d_out)
  k_gemm128  <<<nb_g, 128, 0, stream>>>(h, W2, nullptr, t, N, 0);
  k_selfloop <<<nb_nh4, 256, 0, stream>>>(t, dinv, featL, N);
  k_scatter  <<<nb_ew, 256, 0, stream>>>(src, dst, dinv, t, featL, E);
  k_bias_act <<<nb_nh4, 256, 0, stream>>>(featL, b2, N, 0);

  // conv2u (unlabel): GEMM reads h first, then h is reused as featU
  k_gemm128  <<<nb_g, 128, 0, stream>>>(h, W2u, nullptr, t, N, 0);
  k_selfloop <<<nb_nh4, 256, 0, stream>>>(t, dinv, h, N);       // h := featU
  k_scatter  <<<nb_ew, 256, 0, stream>>>(src, dst, dinv, t, h, E);
  k_bias_act <<<nb_nh4, 256, 0, stream>>>(h, b2u, N, 0);

  // feat = featL + featU -> t (t's previous contents fully consumed above)
  k_add      <<<nb_nh4, 256, 0, stream>>>(featL, h, t, N);

  // heads
  k_gemm128  <<<nb_g, 128, 0, stream>>>(t, Wpca, bpca, opca, N, 1);
  k_gemm128  <<<nb_g, 128, 0, stream>>>(h, Wbce, bbce, obce, N, 1);
  k_head1    <<<nb_nw, 256, 0, stream>>>(featL, Wh1, bh1, out1, N);
}
